// CavityModel_4715874091312
// MI455X (gfx1250) — matrix operand order repaired and checked by reading, not compiled, not run on hardware
//
#include <hip/hip_runtime.h>
#include <hip/hip_bf16.h>

typedef __attribute__((ext_vector_type(16))) _Float16 v16h;
typedef __attribute__((ext_vector_type(8)))  float    v8f;
typedef __attribute__((ext_vector_type(4)))  int      aiv4;

#define NENV 256
#define APE  100
#define NTYP 6
#define GD   18
#define G3   (GD*GD*GD)   // 5832

#if __has_builtin(__builtin_amdgcn_global_load_async_to_lds_b128) && __has_builtin(__builtin_amdgcn_s_wait_asynccnt)
#define USE_ASYNC_LDS 1
// src: global (AS1) int4*, dst: LDS (AS3) int4*, imm offset, imm cpol
__device__ __forceinline__ void async_copy_b128(const void* g, void* l) {
  __builtin_amdgcn_global_load_async_to_lds_b128(
      (__attribute__((address_space(1))) aiv4*)g,
      (__attribute__((address_space(3))) aiv4*)l, 0, 0);
}
#else
#define USE_ASYNC_LDS 0
#endif

// ---------------------------------------------------------------------------
// Utility kernels
// ---------------------------------------------------------------------------
__global__ void zero_f16_kernel(_Float16* p, int n) {
  int i = blockIdx.x * blockDim.x + threadIdx.x;
  if (i < n) p[i] = (_Float16)0.f;
}

__global__ void f32_to_f16_kernel(const float* __restrict__ in, _Float16* __restrict__ out, int n) {
  int i = blockIdx.x * blockDim.x + threadIdx.x;
  if (i < n) out[i] = (_Float16)in[i];
}

// ---------------------------------------------------------------------------
// Gaussian splatting: one block per environment. Separable Gaussian:
// exp(-|p-g|^2 / 2s^2) = ex(i)*ey(j)*ez(k); normalizer = (Sx*Sy*Sz).
// Field (6 x 18^3 fp32 = 140KB) lives entirely in LDS (320KB/WGP on CDNA5).
// Output written as f16 into zero-padded (6,20,20,20) layout for conv1.
// ---------------------------------------------------------------------------
__global__ __launch_bounds__(256) void gaussian_kernel(const float* __restrict__ x,
                                                       _Float16* __restrict__ actpad) {
  __shared__ float field[NTYP * G3];     // 139,968 B
  __shared__ float e[3 * GD];            // ex[18], ey[18], ez[18]
  const int t = threadIdx.x, env = blockIdx.x;
  for (int i = t; i < NTYP * G3; i += 256) field[i] = 0.f;
  __syncthreads();
  const float inv2s2 = 1.0f / (2.0f * 0.6f * 0.6f);
  for (int a = 0; a < APE; ++a) {
    const float* row = x + (size_t)(env * APE + a) * 5;
    const int   typ = (int)row[1];
    const float px = row[2], py = row[3], pz = row[4];
    if (t < 3 * GD) {
      int dim = t / GD, idx = t - dim * GD;
      float c = (float)idx - 8.5f;                 // lin[i] = i - 8.5
      float p = (dim == 0) ? px : ((dim == 1) ? py : pz);
      float d = p - c;
      e[t] = __expf(-d * d * inv2s2);
    }
    __syncthreads();
    float Sx = 0.f, Sy = 0.f, Sz = 0.f;
    #pragma unroll
    for (int i = 0; i < GD; ++i) { Sx += e[i]; Sy += e[GD + i]; Sz += e[2 * GD + i]; }
    const float invS = 1.0f / (Sx * Sy * Sz);
    float* f = field + typ * G3;
    for (int p = t; p < G3; p += 256) {
      int i = p / (GD * GD); int rem = p - i * GD * GD;
      int j = rem / GD;      int k  = rem - j * GD;
      f[p] += e[i] * e[GD + j] * e[2 * GD + k] * invS;   // no race: p owned by thread t
    }
    __syncthreads();
  }
  _Float16* outp = actpad + (size_t)env * NTYP * 20 * 20 * 20;
  for (int p = t; p < NTYP * G3; p += 256) {
    int c = p / G3; int s = p - c * G3;
    int i = s / (GD * GD); int rem = s - i * GD * GD;
    int j = rem / GD;      int k  = rem - j * GD;
    outp[((c * 20 + (i + 1)) * 20 + (j + 1)) * 20 + (k + 1)] = (_Float16)field[p];
  }
}

// ---------------------------------------------------------------------------
// Implicit-GEMM 3D conv via v_wmma_f32_16x16x32_f16, LDS-staged.
// One 8-wave workgroup per (env, 16-out-channel block):
//   1) stage the full input slab (CIN*P^3 f16) + the 16-row weight block into
//      LDS (async-to-LDS on gfx1250, ASYNCcnt-tracked),
//   2) each wave sweeps 16x16 output tiles, building A/B fragments from LDS.
// K-step = one input channel: 27 real taps + 5 zero lanes (weights AND
// activations zeroed -> exact). Fragment packing per CDNA5 ISA layout tables.
// ---------------------------------------------------------------------------
template<int CIN, int P, int DOUT, int COUT>
__global__ __launch_bounds__(256) void conv_wmma_kernel(const _Float16* __restrict__ xin,
                                                        const _Float16* __restrict__ w,   // [COUT][CIN][27] f16
                                                        const float* __restrict__ bias,
                                                        float* __restrict__ out) {        // [N][COUT][S]
  constexpr int S  = DOUT * DOUT * DOUT;
  constexpr int NT = (S + 15) / 16;           // spatial tiles
  constexpr int XV = CIN * P * P * P;         // input slab elements (multiple of 8)
  constexpr int WV = 16 * CIN * 27;           // weight block elements (multiple of 8)
  __shared__ __align__(16) _Float16 lx[XV];
  __shared__ __align__(16) _Float16 lw[WV];

  const int t      = threadIdx.x;
  const int wave   = t >> 5;
  const int lane   = t & 31;
  const int half   = lane >> 4;        // K half-selector
  const int q      = lane & 15;        // A-row (m) / B-col (n)
  const int cobase = blockIdx.x * 16;
  const int env    = blockIdx.y;

  // ---- Stage input slab + weight block into LDS ----
  {
    const _Float16* gx = xin + (size_t)env * XV;
    const _Float16* gw = w   + (size_t)cobase * CIN * 27;
#if USE_ASYNC_LDS
    for (int i = t; i < XV / 8; i += 256) async_copy_b128(gx + i * 8, lx + i * 8);
    for (int i = t; i < WV / 8; i += 256) async_copy_b128(gw + i * 8, lw + i * 8);
    __builtin_amdgcn_s_wait_asynccnt(0);
#else
    for (int i = t; i < XV / 8; i += 256) ((uint4*)lx)[i] = ((const uint4*)gx)[i];
    for (int i = t; i < WV / 8; i += 256) ((uint4*)lw)[i] = ((const uint4*)gw)[i];
#endif
  }
  __syncthreads();

  const _Float16* lwrow = lw + q * CIN * 27;   // A row m = q

  for (int tile = wave; tile < NT; tile += 8) {
    const int  s     = tile * 16 + q;
    const bool valid = (s < S);
    int z = 0, y = 0, xx = 0;
    if (valid) {
      z = s / (DOUT * DOUT); int rem = s - z * DOUT * DOUT;
      y = rem / DOUT;        xx = rem - y * DOUT;
    }
    v8f acc = {};
    for (int ci = 0; ci < CIN; ++ci) {
      // ---- A fragment: W[co=m][k], k = tap index (0..26), 27..31 -> 0
      v16h a;
      #pragma unroll
      for (int v = 0; v < 8; ++v) {
        const int l0 = ((v & 4) << 2) | (half << 3) | ((v & 3) << 1);
        a[2 * v]     = (l0     < 27) ? lwrow[ci * 27 + l0]     : (_Float16)0.f;
        a[2 * v + 1] = (l0 + 1 < 27) ? lwrow[ci * 27 + l0 + 1] : (_Float16)0.f;
      }
      // ---- B fragment: im2col of LDS-resident padded activation, col n = q
      v16h b;
      const _Float16* xc = lx + ci * P * P * P;
      #pragma unroll
      for (int v = 0; v < 8; ++v) {
        #pragma unroll
        for (int u = 0; u < 2; ++u) {
          const int l = (((v & 4) << 2) | (half << 3) | ((v & 3) << 1)) + u;
          _Float16 bv = (_Float16)0.f;
          if (valid && (l < 27)) {
            int dz = l / 9; int rem = l - dz * 9;
            int dy = rem / 3; int dx = rem - dy * 3;
            bv = xc[((z + dz) * P + (y + dy)) * P + (xx + dx)];
          }
          b[2 * v + u] = bv;
        }
      }
      acc = __builtin_amdgcn_wmma_f32_16x16x32_f16(false, a, false, b, (short)0, acc, false, false);
    }
    if (valid) {
      #pragma unroll
      for (int r = 0; r < 8; ++r) {
        const int co = cobase + r + 8 * half;          // D row M = r + 8*half
        out[((size_t)(env * COUT + co)) * S + s] = acc[r] + bias[co];
      }
    }
  }
}

// ---------------------------------------------------------------------------
// 2x2x2 max-pool, stride 2, VALID (floor semantics).
// ---------------------------------------------------------------------------
__global__ void maxpool2_kernel(const float* __restrict__ in, float* __restrict__ out,
                                int C, int Din, int Dout, int total) {
  const int Si = Din * Din * Din, So = Dout * Dout * Dout;
  for (int i = blockIdx.x * blockDim.x + threadIdx.x; i < total; i += gridDim.x * blockDim.x) {
    int n = i / (C * So); int rem = i - n * C * So;
    int c = rem / So;     int sp  = rem - c * So;
    int z = sp / (Dout * Dout); int r2 = sp - z * Dout * Dout;
    int y = r2 / Dout;          int xx = r2 - y * Dout;
    const float* b = in + ((size_t)(n * C + c)) * Si;
    float m = -INFINITY;
    #pragma unroll
    for (int dz = 0; dz < 2; ++dz)
      #pragma unroll
      for (int dy = 0; dy < 2; ++dy)
        #pragma unroll
        for (int dx = 0; dx < 2; ++dx) {
          float v = b[((2 * z + dz) * Din + (2 * y + dy)) * Din + (2 * xx + dx)];
          m = v > m ? v : m;
        }
    out[i] = m;
  }
}

// ---------------------------------------------------------------------------
// BatchNorm stats: one block per channel; population mean/var over (N, S).
// Emits per-channel scale = g*rsqrt(var+eps), shift = b - mean*scale.
// ---------------------------------------------------------------------------
__global__ __launch_bounds__(256) void bnstats_kernel(const float* __restrict__ in,
                                                      const float* __restrict__ g,
                                                      const float* __restrict__ bb,
                                                      float* __restrict__ ss,  // [C][2]
                                                      int C, int S, int N) {
  const int c = blockIdx.x;
  __shared__ float rs[256], rq[256];
  float s = 0.f, q2 = 0.f;
  const int total = N * S;
  for (int i = threadIdx.x; i < total; i += 256) {
    int n = i / S, sp = i - n * S;
    float v = in[((size_t)(n * C + c)) * S + sp];
    s += v; q2 += v * v;
  }
  rs[threadIdx.x] = s; rq[threadIdx.x] = q2;
  __syncthreads();
  for (int o = 128; o > 0; o >>= 1) {
    if (threadIdx.x < o) { rs[threadIdx.x] += rs[threadIdx.x + o]; rq[threadIdx.x] += rq[threadIdx.x + o]; }
    __syncthreads();
  }
  if (threadIdx.x == 0) {
    float mean = rs[0] / total;
    float var  = rq[0] / total - mean * mean;
    float sc   = g[c] * rsqrtf(var + 1e-5f);
    ss[2 * c] = sc; ss[2 * c + 1] = bb[c] - mean * sc;
  }
}

// BN + ReLU, write f16 into (optionally padded) layout for the next conv.
__global__ void bnrelu_pad_kernel(const float* __restrict__ in, const float* __restrict__ ss,
                                  _Float16* __restrict__ outp, int C, int D, int P, int pad, int total) {
  const int S = D * D * D;
  for (int i = blockIdx.x * blockDim.x + threadIdx.x; i < total; i += gridDim.x * blockDim.x) {
    int n = i / (C * S); int rem = i - n * C * S;
    int c = rem / S;     int sp  = rem - c * S;
    int z = sp / (D * D); int r2 = sp - z * D * D;
    int y = r2 / D;       int xx = r2 - y * D;
    float v = in[i] * ss[2 * c] + ss[2 * c + 1];
    v = v > 0.f ? v : 0.f;
    outp[(((size_t)(n * C + c) * P + (z + pad)) * P + (y + pad)) * P + (xx + pad)] = (_Float16)v;
  }
}

// BN + ReLU, f32 output (for the final dense stage).
__global__ void bnrelu_f32_kernel(const float* __restrict__ in, const float* __restrict__ ss,
                                  float* __restrict__ out, int C, int total) {
  int i = blockIdx.x * blockDim.x + threadIdx.x;
  if (i >= total) return;
  int c = i % C;
  float v = in[i] * ss[2 * c] + ss[2 * c + 1];
  out[i] = v > 0.f ? v : 0.f;
}

// ---------------------------------------------------------------------------
// Dense1: D[b][j] = sum_k h[b][k] * W1[j][k] + b1[j] via WMMA (M=batch, N=j, K=64).
// ---------------------------------------------------------------------------
__global__ __launch_bounds__(32) void dense1_wmma_kernel(const _Float16* __restrict__ hf,   // 256x64
                                                         const _Float16* __restrict__ wf,   // 128x64
                                                         const float* __restrict__ bias,
                                                         float* __restrict__ out) {         // 256x128
  const int lane = threadIdx.x;
  const int half = lane >> 4, q = lane & 15;
  const int nb = blockIdx.x * 16;   // feature block
  const int mb = blockIdx.y * 16;   // batch block
  v8f acc = {};
  #pragma unroll
  for (int ks = 0; ks < 2; ++ks) {
    v16h a, b;
    #pragma unroll
    for (int v = 0; v < 8; ++v) {
      const int l0 = ks * 32 + (((v & 4) << 2) | (half << 3) | ((v & 3) << 1));
      a[2 * v]     = hf[(mb + q) * 64 + l0];
      a[2 * v + 1] = hf[(mb + q) * 64 + l0 + 1];
      const int k0 = ks * 32 + (((v & 4) << 2) | (half << 3) | ((v & 3) << 1));
      b[2 * v]     = wf[(nb + q) * 64 + k0];
      b[2 * v + 1] = wf[(nb + q) * 64 + k0 + 1];
    }
    acc = __builtin_amdgcn_wmma_f32_16x16x32_f16(false, a, false, b, (short)0, acc, false, false);
  }
  #pragma unroll
  for (int r = 0; r < 8; ++r) {
    const int m = mb + r + 8 * half;   // batch row
    const int n = nb + q;              // feature col
    out[(size_t)m * 128 + n] = acc[r] + bias[n];
  }
}

// Dense2: tiny (256x21x128) scalar GEMM, full fp32.
__global__ void dense2_kernel(const float* __restrict__ hh, const float* __restrict__ w2,
                              const float* __restrict__ b2, float* __restrict__ out) {
  int i = blockIdx.x * blockDim.x + threadIdx.x;  // n*21 + o
  if (i >= NENV * 21) return;
  int n = i / 21, o = i - n * 21;
  float s = b2[o];
  #pragma unroll 8
  for (int k = 0; k < 128; ++k) s += hh[n * 128 + k] * w2[o * 128 + k];
  out[i] = s;
}

// ---------------------------------------------------------------------------
// Launch
// ---------------------------------------------------------------------------
extern "C" void kernel_launch(void* const* d_in, const int* in_sizes, int n_in,
                              void* d_out, int out_size, void* d_ws, size_t ws_size,
                              hipStream_t stream) {
  const float* x   = (const float*)d_in[0];
  const float* c1w = (const float*)d_in[2];  const float* c1b = (const float*)d_in[3];
  const float* g1  = (const float*)d_in[4];  const float* b1  = (const float*)d_in[5];
  const float* c2w = (const float*)d_in[6];  const float* c2b = (const float*)d_in[7];
  const float* g2  = (const float*)d_in[8];  const float* b2  = (const float*)d_in[9];
  const float* c3w = (const float*)d_in[10]; const float* c3b = (const float*)d_in[11];
  const float* g3  = (const float*)d_in[12]; const float* b3  = (const float*)d_in[13];
  const float* d1w = (const float*)d_in[14]; const float* d1b = (const float*)d_in[15];
  const float* gd  = (const float*)d_in[16]; const float* bd  = (const float*)d_in[17];
  const float* d2w = (const float*)d_in[18]; const float* d2b = (const float*)d_in[19];
  float* out = (float*)d_out;

  // Workspace layout (256B aligned)
  char* ws = (char*)d_ws;
  size_t off = 0;
  auto take = [&](size_t bytes) { char* p = ws + off; off = (off + bytes + 255) & ~(size_t)255; return p; };
  _Float16* actpad  = (_Float16*)take((size_t)NENV * 6 * 8000 * 2);    // 24.6MB (max stage)
  float*    convout = (float*)take((size_t)NENV * 16 * G3 * 4);        // 95.6MB (max stage)
  float*    pooled  = (float*)take((size_t)NENV * 16 * 729 * 4);       // 11.9MB (max stage)
  _Float16* w1f     = (_Float16*)take(16 * 6 * 27 * 2);
  _Float16* w2f     = (_Float16*)take(32 * 16 * 27 * 2);
  _Float16* w3f     = (_Float16*)take(64 * 32 * 27 * 2);
  _Float16* wd1f    = (_Float16*)take(128 * 64 * 2);
  float*    ss      = (float*)take(128 * 2 * 4);
  _Float16* hf16    = (_Float16*)take(NENV * 64 * 2);
  float*    d1out   = (float*)take(NENV * 128 * 4);
  float*    hh      = (float*)take(NENV * 128 * 4);
  (void)ws_size; (void)in_sizes; (void)n_in; (void)out_size;

  const int TB = 256;
  auto cdiv = [](int a, int b) { return (a + b - 1) / b; };

  // Weight prepack (f32 -> f16)
  f32_to_f16_kernel<<<cdiv(16 * 6 * 27, TB), TB, 0, stream>>>(c1w, w1f, 16 * 6 * 27);
  f32_to_f16_kernel<<<cdiv(32 * 16 * 27, TB), TB, 0, stream>>>(c2w, w2f, 32 * 16 * 27);
  f32_to_f16_kernel<<<cdiv(64 * 32 * 27, TB), TB, 0, stream>>>(c3w, w3f, 64 * 32 * 27);
  f32_to_f16_kernel<<<cdiv(128 * 64, TB), TB, 0, stream>>>(d1w, wd1f, 128 * 64);

  // Stage 0: Gaussian fields into padded (6,20,20,20) f16
  {
    int n = NENV * 6 * 8000;
    zero_f16_kernel<<<cdiv(n, TB), TB, 0, stream>>>(actpad, n);
    gaussian_kernel<<<NENV, TB, 0, stream>>>(x, actpad);
  }

  // Stage 1: conv1 (6->16, pad=1, 18^3) -> pool 9^3 -> BN -> ReLU -> f16 (unpadded 9^3)
  conv_wmma_kernel<6, 20, 18, 16><<<dim3(1, NENV), 256, 0, stream>>>(actpad, w1f, c1b, convout);
  {
    int tot = NENV * 16 * 729;
    maxpool2_kernel<<<cdiv(tot, TB), TB, 0, stream>>>(convout, pooled, 16, 18, 9, tot);
    bnstats_kernel<<<16, TB, 0, stream>>>(pooled, g1, b1, ss, 16, 729, NENV);
    bnrelu_pad_kernel<<<cdiv(tot, TB), TB, 0, stream>>>(pooled, ss, actpad, 16, 9, 9, 0, tot);
  }

  // Stage 2: conv2 (16->32, pad=0, 7^3) -> pool 3^3 -> BN -> ReLU -> f16 padded (5^3)
  conv_wmma_kernel<16, 9, 7, 32><<<dim3(2, NENV), 256, 0, stream>>>(actpad, w2f, c2b, convout);
  {
    int tot = NENV * 32 * 27;
    maxpool2_kernel<<<cdiv(tot, TB), TB, 0, stream>>>(convout, pooled, 32, 7, 3, tot);
    bnstats_kernel<<<32, TB, 0, stream>>>(pooled, g2, b2, ss, 32, 27, NENV);
    int np = NENV * 32 * 125;
    zero_f16_kernel<<<cdiv(np, TB), TB, 0, stream>>>(actpad, np);
    bnrelu_pad_kernel<<<cdiv(tot, TB), TB, 0, stream>>>(pooled, ss, actpad, 32, 3, 5, 1, tot);
  }

  // Stage 3: conv3 (32->64, pad=1, 3^3) -> pool 1^3 -> BN -> ReLU -> h (256x64) f16
  conv_wmma_kernel<32, 5, 3, 64><<<dim3(4, NENV), 256, 0, stream>>>(actpad, w3f, c3b, convout);
  {
    int tot = NENV * 64;
    maxpool2_kernel<<<cdiv(tot, TB), TB, 0, stream>>>(convout, pooled, 64, 3, 1, tot);
    bnstats_kernel<<<64, TB, 0, stream>>>(pooled, g3, b3, ss, 64, 1, NENV);
    bnrelu_pad_kernel<<<cdiv(tot, TB), TB, 0, stream>>>(pooled, ss, hf16, 64, 1, 1, 0, tot);
  }

  // Dense1 (WMMA) + BN1d + ReLU
  dense1_wmma_kernel<<<dim3(128 / 16, NENV / 16), 32, 0, stream>>>(hf16, wd1f, d1b, d1out);
  bnstats_kernel<<<128, TB, 0, stream>>>(d1out, gd, bd, ss, 128, 1, NENV);
  bnrelu_f32_kernel<<<cdiv(NENV * 128, TB), TB, 0, stream>>>(d1out, ss, hh, 128, NENV * 128);

  // Dense2 -> output (256x21)
  dense2_kernel<<<cdiv(NENV * 21, TB), TB, 0, stream>>>(hh, d2w, d2b, out);
}
